// HSAAttention_84344567759165
// MI455X (gfx1250) — matrix operand order repaired and checked by reading, not compiled, not run on hardware
//
#include <hip/hip_runtime.h>
#include <hip/hip_bf16.h>

typedef __attribute__((ext_vector_type(16))) _Float16 v16h;
typedef __attribute__((ext_vector_type(8)))  float    v8f;

#define EPI_NONE  0
#define EPI_BIAS  1
#define EPI_SPLAT 2
#define EPI_SCALE 3

__device__ __forceinline__ v8f wmma_f16(v16h a, v16h b, v8f c) {
    // D = A(16x32 f16) * B(32x16 f16) + C(16x16 f32)
    return __builtin_amdgcn_wmma_f32_16x16x32_f16(
        /*neg_a=*/false, a, /*neg_b=*/false, b,
        /*c_mod=*/(short)0, c, /*reuse_a=*/false, /*reuse_b=*/false);
}

// Load a fragment as two contiguous 16B LDS reads (ds_load_b128 pair).
__device__ __forceinline__ v16h ldfrag16(const _Float16* p0, const _Float16* p1) {
    union { float4 f[2]; v16h h; } u;
    u.f[0] = *reinterpret_cast<const float4*>(p0);
    u.f[1] = *reinterpret_cast<const float4*>(p1);
    return u.h;
}

__device__ __forceinline__ void cvt4(_Float16* d, float4 f) {
    d[0] = (_Float16)f.x; d[1] = (_Float16)f.y;
    d[2] = (_Float16)f.z; d[3] = (_Float16)f.w;
}

// ---------------------------------------------------------------------------
// Generic WMMA GEMM: C[M,N] = A[M,K] * B[K,N]  (f16 compute, f32 accumulate)
//   AM=0: A row-major [M,K] (lda=K).   AM=1: A[m][k] = Asrc[k*64 + m] (G^T, M=64)
//   BM=0: B row-major [K,N] (ldb=N).   BM=1: B[k][n] = Bsrc[n*K + k]  (centers^T)
// Double-buffered LDS, register-staged global loads (loads for tile k+1 are in
// flight while tile k computes). One barrier per K-step.
// grid: (N/64, M/128 [>=1], Z) with per-Z element strides.
// ---------------------------------------------------------------------------
template<int AM, int BMODE, int EPI>
__global__ __launch_bounds__(256)
void wmma_gemm_kernel(const float* __restrict__ A, const float* __restrict__ Bsrc,
                      float* __restrict__ C, float* __restrict__ C2,
                      const float* __restrict__ bias,
                      const float* __restrict__ rowterm,   // qq (SPLAT) / rdenom (SCALE)
                      const float* __restrict__ colterm,   // ||c||^2
                      const float* __restrict__ colterm2,  // 1/(2 sigma^2)
                      const float* __restrict__ colterm3,  // amplitudes
                      int M, int N, int K,
                      long strideA, long strideB, long strideC, long strideRow)
{
    const int tid = threadIdx.x;
    const long bz = blockIdx.z;
    A    += bz * strideA;
    Bsrc += bz * strideB;
    C    += bz * strideC;
    if (EPI == EPI_SPLAT) C2 += bz * strideC;
    if (EPI == EPI_SCALE) rowterm += bz * strideRow;

    const int m0 = blockIdx.y * 128;
    const int n0 = blockIdx.x * 64;

    __shared__ __align__(16) _Float16 sA[2][128][40]; // [m][k], padded stride
    __shared__ __align__(16) _Float16 sB[2][64][40];  // [n][k] (B transposed)

    const v8f vzero = {0.f,0.f,0.f,0.f,0.f,0.f,0.f,0.f};
    v8f acc[2][2];
    acc[0][0] = vzero; acc[0][1] = vzero; acc[1][0] = vzero; acc[1][1] = vzero;

    const int wave = tid >> 5;
    const int lane = tid & 31;
    const int wm = (wave & 3) * 32;   // wave row offset in block tile
    const int wn = (wave >> 2) * 32;  // wave col offset
    const int lr = lane & 15;
    const int kh = lane >> 4;

    // G^T path only fills rows 0..63; zero the rest once so WMMA sees no NaNs.
    if (AM == 1) {
        for (int idx = tid; idx < 64 * 40; idx += 256) {
            int r = 64 + idx / 40, c = idx % 40;
            sA[0][r][c] = (_Float16)0.f;
            sA[1][r][c] = (_Float16)0.f;
        }
    }

    float4 aR[4];   // staged A tile (AM==0: 4 passes; AM==1: 2 passes)
    float4 bR[2];   // staged B tile

    auto globalLoadA = [&](int k0) {
        if (AM == 0) {
#pragma unroll
            for (int p = 0; p < 4; ++p) {
                int r = p * 32 + (tid >> 3);
                int c = (tid & 7) * 4;
                aR[p] = *reinterpret_cast<const float4*>(A + (long)(m0 + r) * K + k0 + c);
            }
        } else {
#pragma unroll
            for (int p = 0; p < 2; ++p) {
                int kk = p * 16 + (tid >> 4);
                int m4 = (tid & 15) * 4;     // always < 64, in bounds
                aR[p] = *reinterpret_cast<const float4*>(A + (long)(k0 + kk) * 64 + m4);
            }
        }
    };
    auto globalLoadB = [&](int k0) {
        if (BMODE == 0) {
#pragma unroll
            for (int p = 0; p < 2; ++p) {
                int kk = p * 16 + (tid >> 4);
                int n4 = (tid & 15) * 4;
                bR[p] = *reinterpret_cast<const float4*>(Bsrc + (long)(k0 + kk) * N + n0 + n4);
            }
        } else {
#pragma unroll
            for (int p = 0; p < 2; ++p) {
                int n = p * 32 + (tid >> 3);
                int c = (tid & 7) * 4;
                bR[p] = *reinterpret_cast<const float4*>(Bsrc + (long)(n0 + n) * K + k0 + c);
            }
        }
    };
    auto stage = [&](int buf) {
        if (AM == 0) {
#pragma unroll
            for (int p = 0; p < 4; ++p) {
                int r = p * 32 + (tid >> 3);
                int c = (tid & 7) * 4;
                cvt4(&sA[buf][r][c], aR[p]);   // contiguous 8B f16 store
            }
        } else {
#pragma unroll
            for (int p = 0; p < 2; ++p) {
                int kk = p * 16 + (tid >> 4);
                int m4 = (tid & 15) * 4;
                sA[buf][m4 + 0][kk] = (_Float16)aR[p].x;
                sA[buf][m4 + 1][kk] = (_Float16)aR[p].y;
                sA[buf][m4 + 2][kk] = (_Float16)aR[p].z;
                sA[buf][m4 + 3][kk] = (_Float16)aR[p].w;
            }
        }
        if (BMODE == 0) {
#pragma unroll
            for (int p = 0; p < 2; ++p) {
                int kk = p * 16 + (tid >> 4);
                int n4 = (tid & 15) * 4;
                sB[buf][n4 + 0][kk] = (_Float16)bR[p].x;
                sB[buf][n4 + 1][kk] = (_Float16)bR[p].y;
                sB[buf][n4 + 2][kk] = (_Float16)bR[p].z;
                sB[buf][n4 + 3][kk] = (_Float16)bR[p].w;
            }
        } else {
#pragma unroll
            for (int p = 0; p < 2; ++p) {
                int n = p * 32 + (tid >> 3);
                int c = (tid & 7) * 4;
                cvt4(&sB[buf][n][c], bR[p]);
            }
        }
    };

    const int KT = K / 32;
    globalLoadA(0);
    globalLoadB(0);

    for (int kt = 0; kt < KT; ++kt) {
        const int buf = kt & 1;
        stage(buf);                 // cvt + LDS store of staged registers
        __syncthreads();            // single barrier/iter is safe w/ double buffer
        if (kt + 1 < KT) {          // next tile's loads fly during compute
            globalLoadA((kt + 1) * 32);
            globalLoadB((kt + 1) * 32);
        }
        v16h a0 = ldfrag16(&sA[buf][wm + lr][kh * 8],      &sA[buf][wm + lr][16 + kh * 8]);
        v16h a1 = ldfrag16(&sA[buf][wm + 16 + lr][kh * 8], &sA[buf][wm + 16 + lr][16 + kh * 8]);
        v16h b0 = ldfrag16(&sB[buf][wn + lr][kh * 8],      &sB[buf][wn + lr][16 + kh * 8]);
        v16h b1 = ldfrag16(&sB[buf][wn + 16 + lr][kh * 8], &sB[buf][wn + 16 + lr][16 + kh * 8]);

        acc[0][0] = wmma_f16(a0, b0, acc[0][0]);
        acc[0][1] = wmma_f16(a0, b1, acc[0][1]);
        acc[1][0] = wmma_f16(a1, b0, acc[1][0]);
        acc[1][1] = wmma_f16(a1, b1, acc[1][1]);
    }

    // ---- epilogue + store (C/D layout: lanes 0-15 -> M=v, lanes 16-31 -> M=v+8)
#pragma unroll
    for (int i = 0; i < 2; ++i) {
#pragma unroll
        for (int j = 0; j < 2; ++j) {
            int col = n0 + wn + j * 16 + lr;
#pragma unroll
            for (int v = 0; v < 8; ++v) {
                int row = m0 + wm + i * 16 + kh * 8 + v;
                if (AM == 1 && row >= M) continue;   // only G^T kernel is ragged
                float val = acc[i][j][v];
                if (EPI == EPI_BIAS) {
                    C[(long)row * N + col] = val + bias[col];
                } else if (EPI == EPI_SPLAT) {
                    float d2 = rowterm[row] - 2.0f * val + colterm[col];
                    float g  = __expf(-d2 * colterm2[col]);
                    C [(long)row * N + col] = g;
                    C2[(long)row * N + col] = g * colterm3[col];
                } else if (EPI == EPI_SCALE) {
                    C[(long)row * N + col] = val * rowterm[row];
                } else {
                    C[(long)row * N + col] = val;
                }
            }
        }
    }
}

// ---------------- small helper kernels (wave32 reductions) ------------------

__global__ __launch_bounds__(256)
void rownorm_kernel(const float* __restrict__ q, float* __restrict__ qq, int rows, int D) {
    int wave = threadIdx.x >> 5, lane = threadIdx.x & 31;
    int row = blockIdx.x * 8 + wave;
    if (row >= rows) return;
    const float* p = q + (long)row * D;
    float s = 0.f;
    for (int j = lane; j < D; j += 32) { float x = p[j]; s += x * x; }
#pragma unroll
    for (int off = 16; off; off >>= 1) s += __shfl_xor(s, off, 32);
    if (lane == 0) qq[row] = s;
}

__global__ __launch_bounds__(256)
void prep_kernel(const float* __restrict__ centers, const float* __restrict__ log_scales,
                 float* __restrict__ cc, float* __restrict__ inv2s2, int D) {
    int n = blockIdx.x;
    __shared__ float sh[256];
    float s = 0.f;
    for (int j = threadIdx.x; j < D; j += 256) {
        float x = centers[(long)n * D + j]; s += x * x;
    }
    sh[threadIdx.x] = s; __syncthreads();
    for (int o = 128; o; o >>= 1) {
        if (threadIdx.x < o) sh[threadIdx.x] += sh[threadIdx.x + o];
        __syncthreads();
    }
    if (threadIdx.x == 0) {
        cc[n] = sh[0];
        float sc = __expf(log_scales[n]);
        inv2s2[n] = 0.5f / (sc * sc);
    }
}

__global__ void zero_kernel(float* p, int n) {
    int i = blockIdx.x * 256 + threadIdx.x;
    if (i < n) p[i] = 0.f;
}

__global__ __launch_bounds__(256)
void colsum_kernel(const float* __restrict__ G, float* __restrict__ s, int S) {
    int b = blockIdx.x >> 5, chunk = blockIdx.x & 31;
    int n = threadIdx.x & 63, rq = threadIdx.x >> 6;
    const float* g = G + (long)b * S * 64;
    int base = chunk * 128;
    float p = 0.f;
    for (int it = 0; it < 32; ++it)
        p += g[(long)(base + it * 4 + rq) * 64 + n];
    __shared__ float sh[256];
    sh[threadIdx.x] = p; __syncthreads();
    if (threadIdx.x < 64) {
        float v = sh[threadIdx.x] + sh[threadIdx.x + 64] +
                  sh[threadIdx.x + 128] + sh[threadIdx.x + 192];
        atomicAdd(&s[b * 64 + n], v);
    }
}

__global__ __launch_bounds__(256)
void rdenom_kernel(const float* __restrict__ Ga, const float* __restrict__ s,
                   float* __restrict__ rd, int S) {
    int wave = threadIdx.x >> 5, lane = threadIdx.x & 31;
    int row = blockIdx.x * 8 + wave;
    int b = row / S;
    const float* sr = s + b * 64;
    const float* gr = Ga + (long)row * 64;
    float d = gr[lane] * sr[lane] + gr[lane + 32] * sr[lane + 32];
#pragma unroll
    for (int off = 16; off; off >>= 1) d += __shfl_xor(d, off, 32);
    if (lane == 0) rd[row] = 1.0f / (d + 1e-8f);
}

// ---------------------------------------------------------------------------

extern "C" void kernel_launch(void* const* d_in, const int* in_sizes, int n_in,
                              void* d_out, int out_size, void* d_ws, size_t ws_size,
                              hipStream_t stream) {
    constexpr int B = 4, S = 4096, D = 768, NS = 64;
    constexpr int M = B * S; // 16384 tokens

    const float* query      = (const float*)d_in[0];
    // d_in[1] = key : dead code in the reference output, skipped
    const float* value      = (const float*)d_in[2];
    const float* Wq         = (const float*)d_in[3];
    const float* bq         = (const float*)d_in[4];
    // d_in[5], d_in[6] = Wk, bk : dead
    const float* Wv         = (const float*)d_in[7];
    const float* bv         = (const float*)d_in[8];
    const float* Wo         = (const float*)d_in[9];
    const float* bo         = (const float*)d_in[10];
    const float* centers    = (const float*)d_in[11];
    const float* log_scales = (const float*)d_in[12];
    const float* amplitudes = (const float*)d_in[13];
    float* out = (float*)d_out;

    // workspace partition (256B aligned)
    char* ws = (char*)d_ws;
    size_t off = 0;
    auto alloc = [&](size_t bytes) { void* p = ws + off; off += (bytes + 255) & ~size_t(255); return p; };
    float* q_buf  = (float*)alloc((size_t)M * D * 4);  // q, later reused as attn output
    float* v_buf  = (float*)alloc((size_t)M * D * 4);
    float* G_buf  = (float*)alloc((size_t)M * NS * 4);
    float* Ga_buf = (float*)alloc((size_t)M * NS * 4);
    float* H_buf  = (float*)alloc((size_t)B * NS * D * 4);
    float* s_buf  = (float*)alloc((size_t)B * NS * 4);
    float* qq_buf = (float*)alloc((size_t)M * 4);
    float* rd_buf = (float*)alloc((size_t)M * 4);
    float* cc_buf = (float*)alloc((size_t)NS * 4);
    float* i2_buf = (float*)alloc((size_t)NS * 4);
    if (off > ws_size) return;

    dim3 blk(256);

    // 1) q = query @ Wq + bq        [16384,768] (k-projection is dead -> skipped)
    wmma_gemm_kernel<0,0,EPI_BIAS><<<dim3(D/64, M/128, 1), blk, 0, stream>>>(
        query, Wq, q_buf, nullptr, bq, nullptr, nullptr, nullptr, nullptr,
        M, D, D, 0, 0, 0, 0);

    // 2) v = value @ Wv + bv
    wmma_gemm_kernel<0,0,EPI_BIAS><<<dim3(D/64, M/128, 1), blk, 0, stream>>>(
        value, Wv, v_buf, nullptr, bv, nullptr, nullptr, nullptr, nullptr,
        M, D, D, 0, 0, 0, 0);

    // 3) row norms of q ; center terms
    rownorm_kernel<<<dim3(M/8), blk, 0, stream>>>(q_buf, qq_buf, M, D);
    prep_kernel<<<dim3(NS), blk, 0, stream>>>(centers, log_scales, cc_buf, i2_buf, D);

    // 4) splat: logits = q @ centers^T ; G = exp(-d2/(2s^2)) ; Ga = G*amp
    wmma_gemm_kernel<0,1,EPI_SPLAT><<<dim3(NS/64, M/128, 1), blk, 0, stream>>>(
        q_buf, centers, G_buf, Ga_buf, nullptr, qq_buf, cc_buf, i2_buf, amplitudes,
        M, NS, D, 0, 0, 0, 0);

    // 5) s[b,n] = sum_j G[b,j,n]
    zero_kernel<<<dim3(1), blk, 0, stream>>>(s_buf, B * NS);
    colsum_kernel<<<dim3(B * 32), blk, 0, stream>>>(G_buf, s_buf, S);

    // 6) H[b] = G[b]^T @ v[b]       [64, 768], K = S = 4096
    wmma_gemm_kernel<1,0,EPI_NONE><<<dim3(D/64, 1, B), blk, 0, stream>>>(
        G_buf, v_buf, H_buf, nullptr, nullptr, nullptr, nullptr, nullptr, nullptr,
        NS, D, S, (long)S * NS, (long)S * D, (long)NS * D, 0);

    // 7) rd[row] = 1/(Ga[row]·s[b] + 1e-8)
    rdenom_kernel<<<dim3(M/8), blk, 0, stream>>>(Ga_buf, s_buf, rd_buf, S);

    // 8) attn[b] = (Ga[b] @ H[b]) * rd   -> reuse q_buf
    wmma_gemm_kernel<0,0,EPI_SCALE><<<dim3(D/64, S/128, B), blk, 0, stream>>>(
        Ga_buf, H_buf, q_buf, nullptr, nullptr, rd_buf, nullptr, nullptr, nullptr,
        S, D, NS, (long)S * NS, (long)NS * D, (long)S * D, (long)S);

    // 9) out = attn @ Wo + bo
    wmma_gemm_kernel<0,0,EPI_BIAS><<<dim3(D/64, M/128, 1), blk, 0, stream>>>(
        q_buf, Wo, out, nullptr, bo, nullptr, nullptr, nullptr, nullptr,
        M, D, D, 0, 0, 0, 0);
}
